// GIN_40295383171399
// MI455X (gfx1250) — matrix operand order repaired and checked by reading, not compiled, not run on hardware
//
#include <hip/hip_runtime.h>
#include <hip/hip_bf16.h>

#define N_NODES  100000
#define N_EDGES  1600000
#define CH       128
#define OUTC     64
#define N_GRAPHS 256

typedef __attribute__((ext_vector_type(8)))  unsigned short us8;
typedef __attribute__((ext_vector_type(16))) unsigned short us16;
typedef __attribute__((ext_vector_type(16))) __bf16        v16bf;
typedef __attribute__((ext_vector_type(8)))  float         v8f;

__device__ __forceinline__ unsigned short f2bf(float f) {
    unsigned int u = __float_as_uint(f);
    u += 0x7FFFu + ((u >> 16) & 1u);   // round-to-nearest-even
    return (unsigned short)(u >> 16);
}

// ---------------- simple data movement ----------------
__global__ __launch_bounds__(256) void k_copy4(const float4* __restrict__ in,
                                               float4* __restrict__ out, int n) {
    int i = blockIdx.x * blockDim.x + threadIdx.x;
    if (i < n) out[i] = in[i];
}

__global__ __launch_bounds__(256) void k_zero(float* __restrict__ p, int n) {
    int i = blockIdx.x * blockDim.x + threadIdx.x;
    if (i < n) p[i] = 0.0f;
}

// ---------------- edge scatter-add: one wave (32 lanes) per edge ----------------
// agg must already contain x (self term); adds x[src] into agg[dst].
__global__ __launch_bounds__(256) void k_scatter(const float* __restrict__ X,
                                                 const int* __restrict__ src,
                                                 const int* __restrict__ dst,
                                                 float* __restrict__ agg) {
    int gid  = blockIdx.x * blockDim.x + threadIdx.x;
    int e    = gid >> 5;
    int lane = threadIdx.x & 31;
    if (e >= N_EDGES) return;
    int s = src[e];
    int d = dst[e];
    const float4 v = ((const float4*)(X + (size_t)s * CH))[lane]; // 4 ch per lane
    float* o = agg + (size_t)d * CH + lane * 4;
    atomicAdd(o + 0, v.x);
    atomicAdd(o + 1, v.y);
    atomicAdd(o + 2, v.z);
    atomicAdd(o + 3, v.w);
}

// ---------------- W [K][N] fp32 -> Wt [N][K] bf16 ----------------
__global__ __launch_bounds__(256) void k_wprep(const float* __restrict__ W,
                                               unsigned short* __restrict__ Wt) {
    int i = blockIdx.x * blockDim.x + threadIdx.x;   // i = n*128 + k
    if (i >= CH * CH) return;
    int n = i >> 7, k = i & 127;
    Wt[i] = f2bf(W[k * CH + n]);
}

// ---------------- fused (x+agg)@W + b, ReLU via WMMA bf16 ----------------
// Block = 256 threads = 8 waves; one 16-row M-tile per block; wave w owns cols [16w,16w+16).
__global__ __launch_bounds__(256) void k_gemm(const float* __restrict__ A,
                                              const unsigned short* __restrict__ Wt,
                                              const float* __restrict__ bias,
                                              float* __restrict__ O) {
    __shared__ us8 As8[256];                 // 16 rows x 128 ch, bf16 (4 KB)
    const int t     = threadIdx.x;
    const int tileM = blockIdx.x;

    // Stage A tile -> bf16 LDS: each thread converts 8 consecutive floats.
    {
        const float4* A4 = (const float4*)A;
        int f4 = tileM * 512 + t * 2;
        float4 f0 = A4[f4];
        float4 f1 = A4[f4 + 1];
        us8 s;
        s[0] = f2bf(f0.x); s[1] = f2bf(f0.y); s[2] = f2bf(f0.z); s[3] = f2bf(f0.w);
        s[4] = f2bf(f1.x); s[5] = f2bf(f1.y); s[6] = f2bf(f1.z); s[7] = f2bf(f1.w);
        As8[t] = s;
    }
    __syncthreads();

    const int wave = t >> 5;
    const int lane = t & 31;
    const int half = lane >> 4;
    const int l16  = lane & 15;
    const int row  = l16;                    // A row within tile (lanes 0-15 == 16-31)
    const int col  = wave * 16 + l16;        // output column

    v8f acc = {0.f, 0.f, 0.f, 0.f, 0.f, 0.f, 0.f, 0.f};

#pragma unroll
    for (int kk = 0; kk < 4; ++kk) {         // K = 4 x 32
        // A fragment (ISA 16-bit A 16x32 layout): K = half*8 + {0..7} and 16+half*8+{0..7}
        us8 a0 = As8[row * 16 + kk * 4 + half];
        us8 a1 = As8[row * 16 + kk * 4 + half + 2];
        // B fragment: K = half*8 + {0..7} and 16+half*8+{0..7} of column `col`
        const us8* wp = (const us8*)(Wt + col * CH + kk * 32 + half * 8);
        us8 b0 = wp[0];
        us8 b1 = wp[2];
        us16 au, bu;
#pragma unroll
        for (int i = 0; i < 8; ++i) {
            au[i] = a0[i]; au[i + 8] = a1[i];
            bu[i] = b0[i]; bu[i + 8] = b1[i];
        }
        v16bf af  = __builtin_bit_cast(v16bf, au);
        v16bf bfv = __builtin_bit_cast(v16bf, bu);
        acc = __builtin_amdgcn_wmma_f32_16x16x32_bf16(
            /*neg_a=*/false, af, /*neg_b=*/false, bfv,
            /*c_mod=*/(short)0, acc, /*reuse_a=*/false, /*reuse_b=*/false);
    }

    const float bb = bias[col];
#pragma unroll
    for (int r = 0; r < 8; ++r) {            // D: VGPR r -> row r + 8*half
        float v = acc[r] + bb;
        v = v > 0.f ? v : 0.f;
        O[(size_t)(tileM * 16 + half * 8 + r) * CH + col] = v;
    }
}

// ---------------- mean-pool accumulate: 8 lanes... one float4 per thread ----------------
__global__ __launch_bounds__(256) void k_pool(const float* __restrict__ H,
                                              const int* __restrict__ batch,
                                              float* __restrict__ sums,
                                              float* __restrict__ cnt) {
    int gid  = blockIdx.x * blockDim.x + threadIdx.x;  // node*32 + c4
    int node = gid >> 5;
    int c4   = gid & 31;
    if (node >= N_NODES) return;
    int g = batch[node];
    const float4 v = ((const float4*)(H + (size_t)node * CH))[c4];
    float* o = sums + (size_t)g * CH + c4 * 4;
    atomicAdd(o + 0, v.x);
    atomicAdd(o + 1, v.y);
    atomicAdd(o + 2, v.z);
    atomicAdd(o + 3, v.w);
    if (c4 == 0) atomicAdd(cnt + g, 1.0f);
}

// ---------------- final tiny linear: logits[g][o] ----------------
__global__ __launch_bounds__(256) void k_final(const float* __restrict__ sums,
                                               const float* __restrict__ cnt,
                                               const float* __restrict__ Wf,
                                               const float* __restrict__ bf,
                                               float* __restrict__ out) {
    int i = blockIdx.x * blockDim.x + threadIdx.x;     // g*64 + o
    if (i >= N_GRAPHS * OUTC) return;
    int g = i >> 6, o = i & 63;
    float inv = 1.0f / fmaxf(cnt[g], 1.0f);
    float acc = 0.0f;
#pragma unroll 4
    for (int k = 0; k < CH; ++k)
        acc += sums[g * CH + k] * Wf[k * OUTC + o];
    out[i] = acc * inv + bf[o];
}

extern "C" void kernel_launch(void* const* d_in, const int* in_sizes, int n_in,
                              void* d_out, int out_size, void* d_ws, size_t ws_size,
                              hipStream_t stream) {
    const float* x    = (const float*)d_in[0];
    const int*   eidx = (const int*)d_in[1];
    const int*   batch= (const int*)d_in[2];
    const float* W1 = (const float*)d_in[3];  const float* b1 = (const float*)d_in[4];
    const float* W2 = (const float*)d_in[5];  const float* b2 = (const float*)d_in[6];
    const float* W3 = (const float*)d_in[7];  const float* b3 = (const float*)d_in[8];
    const float* Wf = (const float*)d_in[9];  const float* bf = (const float*)d_in[10];
    const int* src = eidx;
    const int* dst = eidx + N_EDGES;

    float*          buf0 = (float*)d_ws;
    float*          buf1 = buf0 + (size_t)N_NODES * CH;
    unsigned short* Wt   = (unsigned short*)(buf1 + (size_t)N_NODES * CH);
    float*          sums = (float*)(Wt + CH * CH);
    float*          cnt  = sums + (size_t)N_GRAPHS * CH;

    const int nf4   = N_NODES * CH / 4;                 // 3.2M float4
    const int gCopy = (nf4 + 255) / 256;
    const int gScat = (N_EDGES * 32 + 255) / 256;       // wave per edge
    const int gW    = (CH * CH + 255) / 256;
    const int gGemm = N_NODES / 16;                     // 6250 M-tiles
    const int gPool = (N_NODES * 32 + 255) / 256;
    const int gZero = (N_GRAPHS * CH + N_GRAPHS + 255) / 256;
    const int gFin  = (N_GRAPHS * OUTC + 255) / 256;

    const float* layerIn = x;
    const float* Ws[3] = {W1, W2, W3};
    const float* bs[3] = {b1, b2, b3};
    for (int l = 0; l < 3; ++l) {
        k_copy4  <<<gCopy, 256, 0, stream>>>((const float4*)layerIn, (float4*)buf0, nf4);
        k_scatter<<<gScat, 256, 0, stream>>>(layerIn, src, dst, buf0);
        k_wprep  <<<gW,    256, 0, stream>>>(Ws[l], Wt);
        k_gemm   <<<gGemm, 256, 0, stream>>>(buf0, Wt, bs[l], buf1);
        layerIn = buf1;
    }
    k_zero <<<gZero, 256, 0, stream>>>(sums, N_GRAPHS * CH + N_GRAPHS);
    k_pool <<<gPool, 256, 0, stream>>>(buf1, batch, sums, cnt);
    k_final<<<gFin,  256, 0, stream>>>(sums, cnt, Wf, bf, (float*)d_out);
}
